// Qwen3Attention_7876970021296
// MI455X (gfx1250) — matrix operand order repaired and checked by reading, not compiled, not run on hardware
//
#include <hip/hip_runtime.h>
#include <hip/hip_bf16.h>
#include <stdint.h>

// ---------------------------------------------------------------------------
// Qwen3 attention block for MI455X (gfx1250, wave32, WMMA, TDM + async LDS).
// Pipeline: QKV proj (bf16 wmma, LDS-tiled) -> RoPE -> flash attention
// (bf16 wmma, double-buffered K via TDM + V via async copies) -> O proj.
// ---------------------------------------------------------------------------

typedef __attribute__((ext_vector_type(16))) __bf16 bf16x16;
typedef __attribute__((ext_vector_type(8)))  __bf16 bf16x8;
typedef __attribute__((ext_vector_type(8)))  float  f32x8;
typedef __attribute__((ext_vector_type(4)))  unsigned v4u;
typedef __attribute__((ext_vector_type(8)))  int      v8i;
typedef __attribute__((ext_vector_type(4)))  int      v4i;

#define HEADS     32
#define KV_HEADS  8
#define HEAD_DIM  128
#define SEQ       2048
#define BATCH     2
#define DMODEL    4096
#define TOKENS    (BATCH * SEQ)
#define KV_STRIDE (KV_HEADS * HEAD_DIM)      // 1024

static __device__ __forceinline__ f32x8
wmma_bf16(const bf16x16& a, const bf16x16& b, const f32x8& c) {
    return __builtin_amdgcn_wmma_f32_16x16x32_bf16(
        false, a, false, b, (short)0, c, false, false);
}

static __device__ __forceinline__ bf16x16 cat8(bf16x8 lo, bf16x8 hi) {
    return __builtin_shufflevector(lo, hi, 0, 1, 2, 3, 4, 5, 6, 7,
                                           8, 9, 10, 11, 12, 13, 14, 15);
}

// ---- async global -> LDS, 16B per lane (ASYNCcnt) ----
static __device__ __forceinline__ void async_copy_b128(const void* gaddr, void* lds) {
    const unsigned loff = (unsigned)(uintptr_t)lds;
    asm volatile("global_load_async_to_lds_b128 %0, %1, off"
                 :: "v"(loff), "v"(gaddr) : "memory");
}
static __device__ __forceinline__ void async_wait0() {
    asm volatile("s_wait_asynccnt 0" ::: "memory");
}

// ---- Tensor Data Mover: 2D bf16 tile, global -> LDS (TENSORcnt) ----
// D# per ISA 8.3/8.4: group0 {count, lds_addr, global_addr, type=2},
// group1 {data_size=1 (2B), tensor dims, tile dims, dim0 stride}.
// clang-23 lane: 6-arg builtin (g0, g1, g2, g3, g4, cpol).
static __device__ __forceinline__ void tdm_load_2d_bf16(unsigned lds_off, const void* gbase,
                                                        unsigned tile_d0, unsigned tile_d1,
                                                        unsigned stride_d0) {
#if __has_builtin(__builtin_amdgcn_tensor_load_to_lds)
    const unsigned long long ga = (unsigned long long)(uintptr_t)gbase;
    v4u g0;
    g0[0] = 1u;                                            // count=1 (valid user D#)
    g0[1] = lds_off;                                       // lds_addr
    g0[2] = (unsigned)(ga & 0xFFFFFFFFu);                  // global_addr[31:0]
    g0[3] = (unsigned)((ga >> 32) & 0x01FFFFFFu) | (2u << 30); // [120:96] + type=2
    v8i g1;
    g1[0] = 0x00010000;                                    // data_size=1 -> 2 bytes
    g1[1] = (int)(tile_d0 << 16);                          // tensor_dim0[15:0] @ [63:48]
    g1[2] = (int)((tile_d0 >> 16) | (tile_d1 << 16));      // tensor_dim0 hi | tensor_dim1 lo
    g1[3] = (int)((tile_d1 >> 16) | (tile_d0 << 16));      // tensor_dim1 hi | tile_dim0
    g1[4] = (int)tile_d1;                                  // tile_dim1 (tile_dim2 = 0)
    g1[5] = (int)stride_d0;                                // tensor_dim0_stride[31:0]
    g1[6] = 0;
    g1[7] = 0;
    const v4i z4 = {0, 0, 0, 0};                           // groups 2/3 unused (2D)
    const v8i z8 = {0, 0, 0, 0, 0, 0, 0, 0};
    __builtin_amdgcn_tensor_load_to_lds(g0, g1, z4, z4, z8, 0);
#else
    (void)lds_off; (void)gbase; (void)tile_d0; (void)tile_d1; (void)stride_d0;
#endif
}
static __device__ __forceinline__ void tdm_wait0() {
#if __has_builtin(__builtin_amdgcn_s_wait_tensorcnt)
    __builtin_amdgcn_s_wait_tensorcnt(0);
#endif
}

// ---------------------------------------------------------------------------
// GEMM: C[M,N] = A[M,K](f32) @ W[N,K]^T(f32), bf16 WMMA, f32 accumulate.
// Block 256 thr = 8 waves (2M x 4N), wave tile 16x64, block tile 32x256.
// transC=1 stores C transposed (used for V so attention reads Vt).
// ---------------------------------------------------------------------------
template <bool BF16OUT>
__global__ __launch_bounds__(256) void gemm_xWT(const float* __restrict__ A,
                                                const float* __restrict__ W,
                                                void* __restrict__ Cv,
                                                int M, int N, int K, int transC) {
    __shared__ __bf16 Abuf[32][32];    // 2 KB
    __shared__ __bf16 Wbuf[256][32];   // 16 KB

    const int tid  = threadIdx.x;
    const int lane = tid & 31;
    const int wave = tid >> 5;
    const int wm   = wave >> 2;
    const int wn   = wave & 3;
    const int r    = lane & 15;
    const int kh   = lane >> 4;
    const int mblk = blockIdx.y * 32;
    const int nblk = blockIdx.x * 256;

    const int arow = tid >> 3;
    const int akq  = (tid & 7) * 4;

    f32x8 acc[4] = {};

    for (int k0 = 0; k0 < K; k0 += 32) {
        __syncthreads();
        {
            float4 f = *(const float4*)(A + (size_t)(mblk + arow) * K + k0 + akq);
            Abuf[arow][akq + 0] = (__bf16)f.x;
            Abuf[arow][akq + 1] = (__bf16)f.y;
            Abuf[arow][akq + 2] = (__bf16)f.z;
            Abuf[arow][akq + 3] = (__bf16)f.w;
        }
        {
            const float* wr = W + (size_t)(nblk + tid) * K + k0;
#pragma unroll
            for (int q = 0; q < 8; ++q) {
                float4 f = *(const float4*)(wr + q * 4);
                Wbuf[tid][q * 4 + 0] = (__bf16)f.x;
                Wbuf[tid][q * 4 + 1] = (__bf16)f.y;
                Wbuf[tid][q * 4 + 2] = (__bf16)f.z;
                Wbuf[tid][q * 4 + 3] = (__bf16)f.w;
            }
        }
        __syncthreads();

        bf16x16 a = cat8(*(const bf16x8*)&Abuf[wm * 16 + r][kh * 8],
                         *(const bf16x8*)&Abuf[wm * 16 + r][16 + kh * 8]);
#pragma unroll
        for (int nt = 0; nt < 4; ++nt) {
            const __bf16* wrow = &Wbuf[wn * 64 + nt * 16 + r][kh * 8];
            bf16x16 bm = cat8(*(const bf16x8*)wrow, *(const bf16x8*)(wrow + 16));
            acc[nt] = wmma_bf16(a, bm, acc[nt]);
        }
    }

#pragma unroll
    for (int nt = 0; nt < 4; ++nt) {
#pragma unroll
        for (int j = 0; j < 8; ++j) {
            const int row = mblk + wm * 16 + j + 8 * kh;
            const int col = nblk + wn * 64 + nt * 16 + r;
            const size_t off = transC ? ((size_t)col * M + row)
                                      : ((size_t)row * N + col);
            if (BF16OUT) ((__bf16*)Cv)[off] = (__bf16)acc[nt][j];
            else         ((float*)Cv)[off]  = acc[nt][j];
        }
    }
}

// ---------------------------------------------------------------------------
// In-place interleaved-pair RoPE on bf16 [T, heads, 128].
// ---------------------------------------------------------------------------
__global__ __launch_bounds__(256) void rope_kernel(__bf16* __restrict__ x,
                                                   const float* __restrict__ freqs,
                                                   const int* __restrict__ poff,
                                                   int heads) {
    const int idx   = blockIdx.x * blockDim.x + threadIdx.x;
    const int total = TOKENS * heads * (HEAD_DIM / 2);
    if (idx >= total) return;
    const int i = idx & 63;
    const int h = (idx >> 6) % heads;
    const int t = idx / (64 * heads);
    const int s = t & (SEQ - 1);

    const float ang = freqs[(size_t)(s + *poff) * (HEAD_DIM / 2) + i];
    const float c  = __cosf(ang);
    const float sn = __sinf(ang);

    const size_t p  = (size_t)t * (heads * HEAD_DIM) + h * HEAD_DIM + 2 * i;
    const float  x1 = (float)x[p];
    const float  x2 = (float)x[p + 1];
    x[p]     = (__bf16)(x1 * c - x2 * sn);
    x[p + 1] = (__bf16)(x1 * sn + x2 * c);
}

// ---------------------------------------------------------------------------
// Flash attention (no causal mask), GQA group = 4.
// 8 waves/block share one (b, kv-head). Double-buffered chunk staging:
//   K tile  (32x128 bf16) via Tensor Data Mover, issued by wave 0,
//   Vt tile (128x32 bf16) via per-thread async global->LDS b128 copies.
// Copies for chunk ch+1 overlap WMMA compute on chunk ch.
// ---------------------------------------------------------------------------
__global__ __launch_bounds__(256) void attn_kernel(const __bf16* __restrict__ Q,
                                                   const __bf16* __restrict__ K,
                                                   const __bf16* __restrict__ Vt,
                                                   float* __restrict__ O) {
    __shared__ __bf16 Kbuf[2][32][128];      // 2 x 8 KB [kv][hd]
    __shared__ __bf16 Vbuf[2][128][32];      // 2 x 8 KB [hd][kv]
    __shared__ __bf16 Ptile[8][16 * 32];     // 8 KB per-wave P

    const int tid  = threadIdx.x;
    const int lane = tid & 31;
    const int wave = tid >> 5;
    const int tile = blockIdx.x & 15;
    const int bh   = blockIdx.x >> 4;
    const int h    = bh & (HEADS - 1);
    const int b    = bh >> 5;
    const int kvh  = h >> 2;
    const int r    = lane & 15;
    const int kh   = lane >> 4;
    const int s0   = tile * 128 + wave * 16;
    const float scale = 0.08838834764831845f;   // 1/sqrt(128)
    const int NCH = SEQ / 32;

    const __bf16* Kg = K  + (size_t)(b * SEQ) * KV_STRIDE + kvh * HEAD_DIM;
    const __bf16* Vg = Vt + (size_t)(kvh * HEAD_DIM) * TOKENS + b * SEQ;

    // V staging coords: 128 rows x 32 elems, 2 x b128 per thread
    const int vrow = tid >> 1;
    const int vseg = (tid & 1) * 8;

    // ---- preload 4 Q A-fragments ----
    bf16x16 qf[4];
    const __bf16* qrow =
        Q + (size_t)(b * SEQ + s0 + r) * (HEADS * HEAD_DIM) + h * HEAD_DIM;
#pragma unroll
    for (int c = 0; c < 4; ++c)
        qf[c] = cat8(*(const bf16x8*)(qrow + c * 32 + kh * 8),
                     *(const bf16x8*)(qrow + c * 32 + 16 + kh * 8));

    float mrun[8], lrun[8];
    f32x8 accO[8] = {};
#pragma unroll
    for (int j = 0; j < 8; ++j) { mrun[j] = -1e30f; lrun[j] = 0.0f; }

    // ---- prologue: stage chunk 0 into buffer 0 ----
    if (wave == 0)
        tdm_load_2d_bf16((unsigned)(uintptr_t)&Kbuf[0][0][0],
                         Kg, HEAD_DIM, 32, KV_STRIDE);
    async_copy_b128(Vg + (size_t)vrow * TOKENS + vseg,      &Vbuf[0][vrow][vseg]);
    async_copy_b128(Vg + (size_t)vrow * TOKENS + vseg + 16, &Vbuf[0][vrow][vseg + 16]);
    async_wait0();
    if (wave == 0) tdm_wait0();
    __syncthreads();

    for (int ch = 0; ch < NCH; ++ch) {
        const int kv0 = ch * 32;
        const int cur = ch & 1;
        const int nxt = cur ^ 1;

        // ---- kick off copies for next chunk (overlap with compute) ----
        if (ch + 1 < NCH) {
            if (wave == 0)
                tdm_load_2d_bf16((unsigned)(uintptr_t)&Kbuf[nxt][0][0],
                                 Kg + (size_t)(kv0 + 32) * KV_STRIDE,
                                 HEAD_DIM, 32, KV_STRIDE);
            async_copy_b128(Vg + (size_t)vrow * TOKENS + kv0 + 32 + vseg,
                            &Vbuf[nxt][vrow][vseg]);
            async_copy_b128(Vg + (size_t)vrow * TOKENS + kv0 + 32 + vseg + 16,
                            &Vbuf[nxt][vrow][vseg + 16]);
        }

        // ---- scores: Q(16x128) @ K^T -> two 16-key groups ----
        f32x8 sc0 = {}, sc1 = {};
#pragma unroll
        for (int c = 0; c < 4; ++c) {
            const __bf16* k0p = &Kbuf[cur][r][c * 32 + kh * 8];
            bf16x16 kf0 = cat8(*(const bf16x8*)k0p, *(const bf16x8*)(k0p + 16));
            sc0 = wmma_bf16(qf[c], kf0, sc0);
            const __bf16* k1p = &Kbuf[cur][16 + r][c * 32 + kh * 8];
            bf16x16 kf1 = cat8(*(const bf16x8*)k1p, *(const bf16x8*)(k1p + 16));
            sc1 = wmma_bf16(qf[c], kf1, sc1);
        }

        // ---- online softmax (rows in VGPR j; reduce within 16-lane half) ----
        float pfac[8];
#pragma unroll
        for (int j = 0; j < 8; ++j) {
            const float a0 = sc0[j] * scale;
            const float a1 = sc1[j] * scale;
            float mx = fmaxf(a0, a1);
#pragma unroll
            for (int m = 1; m < 16; m <<= 1) mx = fmaxf(mx, __shfl_xor(mx, m, 32));
            const float mn    = fmaxf(mrun[j], mx);
            const float alpha = __expf(mrun[j] - mn);
            const float e0    = __expf(a0 - mn);
            const float e1    = __expf(a1 - mn);
            float rs = e0 + e1;
#pragma unroll
            for (int m = 1; m < 16; m <<= 1) rs += __shfl_xor(rs, m, 32);
            lrun[j] = lrun[j] * alpha + rs;
            mrun[j] = mn;
            pfac[j] = alpha;
            Ptile[wave][(j + 8 * kh) * 32 + r]      = (__bf16)e0;
            Ptile[wave][(j + 8 * kh) * 32 + 16 + r] = (__bf16)e1;
        }
#pragma unroll
        for (int nt = 0; nt < 8; ++nt)
#pragma unroll
            for (int j = 0; j < 8; ++j) accO[nt][j] *= pfac[j];

        // ---- P A-fragment from wave-private LDS ----
        bf16x16 pf = cat8(*(const bf16x8*)&Ptile[wave][r * 32 + kh * 8],
                          *(const bf16x8*)&Ptile[wave][r * 32 + 16 + kh * 8]);

        // ---- PV: P(16x32) @ V(32x128) ----
#pragma unroll
        for (int nt = 0; nt < 8; ++nt) {
            const __bf16* vp = &Vbuf[cur][nt * 16 + r][kh * 8];
            bf16x16 vf = cat8(*(const bf16x8*)vp, *(const bf16x8*)(vp + 16));
            accO[nt] = wmma_bf16(pf, vf, accO[nt]);
        }

        // ---- ensure next chunk landed; all waves done with current ----
        if (ch + 1 < NCH) {
            async_wait0();
            if (wave == 0) tdm_wait0();
        }
        __syncthreads();
    }

    // ---- normalize and write f32 [T, H*HD] ----
#pragma unroll
    for (int nt = 0; nt < 8; ++nt) {
#pragma unroll
        for (int j = 0; j < 8; ++j) {
            const int t = b * SEQ + s0 + j + 8 * kh;
            O[(size_t)t * (HEADS * HEAD_DIM) + h * HEAD_DIM + nt * 16 + r] =
                accO[nt][j] / lrun[j];
        }
    }
}

// ---------------------------------------------------------------------------
extern "C" void kernel_launch(void* const* d_in, const int* in_sizes, int n_in,
                              void* d_out, int out_size, void* d_ws, size_t ws_size,
                              hipStream_t stream) {
    const float* hs    = (const float*)d_in[0];
    const float* freqs = (const float*)d_in[1];
    const float* Wq    = (const float*)d_in[2];
    const float* Wk    = (const float*)d_in[3];
    const float* Wv    = (const float*)d_in[4];
    const float* Wo    = (const float*)d_in[5];
    const int*   poff  = (const int*)d_in[6];
    float*       out   = (float*)d_out;

    // workspace carve: Q bf16 32M | K bf16 8M | Vt bf16 8M | attnO f32 64M
    char* ws = (char*)d_ws;
    __bf16* Qb  = (__bf16*)ws;
    __bf16* Kb  = (__bf16*)(ws + (size_t)TOKENS * DMODEL * 2);
    __bf16* Vtb = (__bf16*)(ws + (size_t)TOKENS * DMODEL * 2 + (size_t)TOKENS * 1024 * 2);
    float*  AO  = (float*)(ws + (size_t)TOKENS * DMODEL * 2 + (size_t)TOKENS * 1024 * 4);

    const dim3 blk(256);

    gemm_xWT<true><<<dim3(DMODEL / 256, TOKENS / 32), blk, 0, stream>>>(hs, Wq, Qb,  TOKENS, DMODEL, DMODEL, 0);
    gemm_xWT<true><<<dim3(1024 / 256,   TOKENS / 32), blk, 0, stream>>>(hs, Wk, Kb,  TOKENS, 1024,   DMODEL, 0);
    gemm_xWT<true><<<dim3(1024 / 256,   TOKENS / 32), blk, 0, stream>>>(hs, Wv, Vtb, TOKENS, 1024,   DMODEL, 1);

    rope_kernel<<<(TOKENS * HEADS * 64) / 256,    blk, 0, stream>>>(Qb, freqs, poff, HEADS);
    rope_kernel<<<(TOKENS * KV_HEADS * 64) / 256, blk, 0, stream>>>(Kb, freqs, poff, KV_HEADS);

    attn_kernel<<<BATCH * HEADS * (SEQ / 128), blk, 0, stream>>>(Qb, Kb, Vtb, AO);

    gemm_xWT<false><<<dim3(DMODEL / 256, TOKENS / 32), blk, 0, stream>>>(AO, Wo, out, TOKENS, DMODEL, DMODEL, 0);
}
